// MultiHeadAttention_26963804684369
// MI455X (gfx1250) — matrix operand order repaired and checked by reading, not compiled, not run on hardware
//
#include <hip/hip_runtime.h>
#include <hip/hip_bf16.h>

#define MDIM 2048
#define SEQDIM 4096
#define NHEAD 16
#define HDIM 128

typedef __attribute__((ext_vector_type(16))) __bf16 v16bf;
typedef __attribute__((ext_vector_type(8)))  float  v8f;
typedef __attribute__((ext_vector_type(4)))  int    v4i;

union Frag {
    v16bf v;
    unsigned u[8];
};

#ifndef __has_builtin
#define __has_builtin(x) 0
#endif

#if __has_builtin(__builtin_amdgcn_global_load_async_to_lds_b128)
#define HAVE_ASYNC_LDS 1
#else
#define HAVE_ASYNC_LDS 0
#endif

#define GLOBAL_AS __attribute__((address_space(1)))
#define LDS_AS    __attribute__((address_space(3)))

// 16-byte global -> LDS copy. Async (ASYNCcnt) when available, sync fallback.
__device__ __forceinline__ void async_copy16(void* lds, const void* g) {
#if HAVE_ASYNC_LDS
    // Builtin signature (from diagnostics): (v4i AS1*, v4i AS3*, imm offset, imm cpol)
    // AS1 addresses == flat addresses for global; flat[31:0] == LDS offset (ISA 10.2)
    __builtin_amdgcn_global_load_async_to_lds_b128(
        (GLOBAL_AS v4i*)(uintptr_t)g,
        (LDS_AS v4i*)(uintptr_t)lds, 0, 0);
#else
    *(uint4*)lds = *(const uint4*)g;
#endif
}

__device__ __forceinline__ void async_wait() {
#if HAVE_ASYNC_LDS
#if __has_builtin(__builtin_amdgcn_s_wait_asynccnt)
    __builtin_amdgcn_s_wait_asynccnt(0);
#else
    asm volatile("s_wait_asynccnt 0" ::: "memory");
#endif
#endif
}

__device__ __forceinline__ v8f wmma_bf16(v16bf a, v16bf b, v8f c) {
    return __builtin_amdgcn_wmma_f32_16x16x32_bf16(
        false, a, false, b, (short)0, c, false, false);
}

// A-fragment K index (16-bit 16x32 A): lanes 0-15 K {0..7,16..23}; lanes 16-31 K {8..15,24..31}
__device__ __forceinline__ int a_kidx(int v, int half) {
    return (v < 4) ? (half * 8 + 2 * v) : (16 + half * 8 + 2 * (v - 4));
}
// B-fragment K index (16-bit 32x16 B): lane = N; lanes 0-15 K {0..7,16..23}; lanes 16-31 K {8..15,24..31}
__device__ __forceinline__ int b_kidx(int v, int half) {
    return (v < 4) ? (half * 8 + 2 * v) : (16 + half * 8 + 2 * (v - 4));
}

// ---------------------------------------------------------------------------
// f32 -> bf16 elementwise (4 elems/thread)
// ---------------------------------------------------------------------------
__global__ __launch_bounds__(256) void cvt_bf16(const float* __restrict__ in,
                                                __bf16* __restrict__ out) {
    size_t i = ((size_t)blockIdx.x * 256 + threadIdx.x) * 4;
    float4 f = *(const float4*)(in + i);
    out[i + 0] = (__bf16)f.x;
    out[i + 1] = (__bf16)f.y;
    out[i + 2] = (__bf16)f.z;
    out[i + 3] = (__bf16)f.w;
}

// ---------------------------------------------------------------------------
// W[K][N] f32 -> Wt[N][K] bf16 (32x32 LDS tile transpose)
// ---------------------------------------------------------------------------
__global__ __launch_bounds__(256) void cvt_transpose_bf16(const float* __restrict__ W,
                                                          __bf16* __restrict__ Wt) {
    __shared__ __bf16 tile[32][33];
    const int bn = blockIdx.x * 32;   // N
    const int bk = blockIdx.y * 32;   // K
    const int tx = threadIdx.x & 31;
    const int ty = threadIdx.x >> 5;  // 0..7
#pragma unroll
    for (int r = 0; r < 4; ++r) {
        int k = ty + r * 8;
        tile[k][tx] = (__bf16)W[(size_t)(bk + k) * MDIM + bn + tx];
    }
    __syncthreads();
#pragma unroll
    for (int r = 0; r < 4; ++r) {
        int n = ty + r * 8;
        Wt[(size_t)(bn + n) * MDIM + bk + tx] = tile[tx][n];
    }
}

// ---------------------------------------------------------------------------
// GEMM (NT): Out[M,N](bf16) = A[M,K](bf16) @ Bt[N,K](bf16)^T + bias
// Block 128x128, 8 waves (2Mx4N), wave 64x32, K step 32, double-buffered LDS
// fed by async global->LDS copies.
// ---------------------------------------------------------------------------
__global__ __launch_bounds__(256) void gemm_bf16_nt(
    const __bf16* __restrict__ A,
    const __bf16* __restrict__ Bt,
    const float* __restrict__ bias,
    __bf16* __restrict__ Out)
{
    __shared__ __bf16 As[2][128][40];   // row stride 80B (16B multiple), banks conflict-free
    __shared__ __bf16 Bs[2][128][40];

    const int tid   = threadIdx.x;
    const int wave  = tid >> 5;
    const int lane  = tid & 31;
    const int l16   = lane & 15;
    const int half  = lane >> 4;
    const int waveM = wave >> 2;
    const int waveN = wave & 3;
    const int blockM = blockIdx.y * 128;
    const int blockN = blockIdx.x * 128;

    // Per-thread tile-copy coords: 256 threads x (1 A + 1 B) 16B segments
    const int cpRow = tid >> 1;          // 0..127
    const int cpSeg = (tid & 1) << 3;    // 0 or 8 (bf16 elems)
    const __bf16* gA = A  + (size_t)(blockM + cpRow) * MDIM + cpSeg;
    const __bf16* gB = Bt + (size_t)(blockN + cpRow) * MDIM + cpSeg;

    v8f c[4][2];
#pragma unroll
    for (int mi = 0; mi < 4; ++mi)
#pragma unroll
        for (int ni = 0; ni < 2; ++ni)
            c[mi][ni] = (v8f){0.f, 0.f, 0.f, 0.f, 0.f, 0.f, 0.f, 0.f};

    auto stage_load = [&](int s, int k0) {
        async_copy16(&As[s][cpRow][cpSeg], gA + k0);
        async_copy16(&Bs[s][cpRow][cpSeg], gB + k0);
    };
    auto compute = [&](int s) {
        Frag a[4], b[2];
#pragma unroll
        for (int mi = 0; mi < 4; ++mi) {
            int row = waveM * 64 + mi * 16 + l16;
#pragma unroll
            for (int v = 0; v < 8; ++v)
                a[mi].u[v] = *(const unsigned*)&As[s][row][a_kidx(v, half)];
        }
#pragma unroll
        for (int ni = 0; ni < 2; ++ni) {
            int col = waveN * 32 + ni * 16 + l16;
#pragma unroll
            for (int v = 0; v < 8; ++v)
                b[ni].u[v] = *(const unsigned*)&Bs[s][col][b_kidx(v, half)];
        }
#pragma unroll
        for (int mi = 0; mi < 4; ++mi)
#pragma unroll
            for (int ni = 0; ni < 2; ++ni)
                c[mi][ni] = wmma_bf16(a[mi].v, b[ni].v, c[mi][ni]);
    };

    stage_load(0, 0);
    async_wait();
    __syncthreads();

    int cur = 0;
    for (int k0 = 32; k0 < MDIM; k0 += 32) {
        stage_load(cur ^ 1, k0);   // prefetch next stage (async, overlaps WMMA)
        compute(cur);
        async_wait();
        __syncthreads();
        cur ^= 1;
    }
    compute(cur);

    // Epilogue: C layout row M = r + 8*half, col N = lane%16
#pragma unroll
    for (int mi = 0; mi < 4; ++mi)
#pragma unroll
        for (int ni = 0; ni < 2; ++ni) {
            int col  = blockN + waveN * 32 + ni * 16 + l16;
            float bv = bias[col];
#pragma unroll
            for (int r = 0; r < 8; ++r) {
                int row = blockM + waveM * 64 + mi * 16 + r + 8 * half;
                Out[(size_t)row * MDIM + col] = (__bf16)(c[mi][ni][r] + bv);
            }
        }
}

// ---------------------------------------------------------------------------
// Flash attention: block = (head, 128-query tile); 8 waves x 16 q-rows.
// K tile via async global->LDS; V transposed through VGPRs; online softmax.
// ---------------------------------------------------------------------------
__global__ __launch_bounds__(256) void attn_flash_bf16(
    const __bf16* __restrict__ Qb,
    const __bf16* __restrict__ Kb,
    const __bf16* __restrict__ Vb,
    float* __restrict__ Out)
{
    __shared__ __bf16 Ks[64][136];     // [key][d], row stride 272B (16B multiple)
    __shared__ __bf16 Vts[128][68];    // [d][key]
    __shared__ __bf16 Ps[8][16][68];   // per-wave P tile [m][key]

    const int tid  = threadIdx.x;
    const int wave = tid >> 5;
    const int lane = tid & 31;
    const int l16  = lane & 15;
    const int half = lane >> 4;
    const int h    = blockIdx.y;
    const int qb   = blockIdx.x;
    const int qRow0 = qb * 128 + wave * 16;
    const size_t colBase = (size_t)h * HDIM;

    Frag aQ[4];
    {
        int row = qRow0 + l16;
        const __bf16* qp = Qb + (size_t)row * MDIM + colBase;
#pragma unroll
        for (int kk = 0; kk < 4; ++kk)
#pragma unroll
            for (int v = 0; v < 8; ++v)
                aQ[kk].u[v] = *(const unsigned*)(qp + kk * 32 + a_kidx(v, half));
    }

    v8f o[8];
#pragma unroll
    for (int nd = 0; nd < 8; ++nd)
        o[nd] = (v8f){0.f, 0.f, 0.f, 0.f, 0.f, 0.f, 0.f, 0.f};
    float m_run[8], l_run[8];
#pragma unroll
    for (int r = 0; r < 8; ++r) { m_run[r] = -1e30f; l_run[r] = 0.f; }

    const float scale = 0.0883883476483184f;   // 1/sqrt(128)

    for (int kb = 0; kb < SEQDIM / 64; ++kb) {
        __syncthreads();                        // prior reads of Ks/Vts/Ps done
        const int keyBase = kb * 64;
        // K tile 64x128 bf16: async straight into LDS
#pragma unroll
        for (int j = 0; j < 4; ++j) {
            int idx = tid + j * 256;
            int t   = idx >> 4;
            int c8  = (idx & 15) << 3;
            async_copy16(&Ks[t][c8],
                         Kb + (size_t)(keyBase + t) * MDIM + colBase + c8);
        }
        // V tile transposed through VGPRs: Vts[d][t]
#pragma unroll
        for (int j = 0; j < 4; ++j) {
            int idx = tid + j * 256;
            int t   = idx >> 4;
            int c8  = (idx & 15) << 3;
            uint4 val = *(const uint4*)(Vb + (size_t)(keyBase + t) * MDIM + colBase + c8);
            const __bf16* e = (const __bf16*)&val;
#pragma unroll
            for (int q = 0; q < 8; ++q) Vts[c8 + q][t] = e[q];
        }
        async_wait();
        __syncthreads();

        // Scores: S[16x64] = Q @ K^T
        v8f s[4];
#pragma unroll
        for (int ni = 0; ni < 4; ++ni)
            s[ni] = (v8f){0.f, 0.f, 0.f, 0.f, 0.f, 0.f, 0.f, 0.f};
#pragma unroll
        for (int ni = 0; ni < 4; ++ni) {
            int n = ni * 16 + l16;
#pragma unroll
            for (int kk = 0; kk < 4; ++kk) {
                Frag bK;
#pragma unroll
                for (int v = 0; v < 8; ++v)
                    bK.u[v] = *(const unsigned*)&Ks[n][kk * 32 + b_kidx(v, half)];
                s[ni] = wmma_bf16(aQ[kk].v, bK.v, s[ni]);
            }
        }

        // Online softmax (row M = r + 8*half; 16-lane halves hold the 16 cols)
        float p[4][8];
#pragma unroll
        for (int r = 0; r < 8; ++r) {
            float mx = -1e30f;
#pragma unroll
            for (int ni = 0; ni < 4; ++ni) {
                float vv = s[ni][r] * scale;
                p[ni][r] = vv;
                mx = fmaxf(mx, vv);
            }
#pragma unroll
            for (int off = 8; off >= 1; off >>= 1)
                mx = fmaxf(mx, __shfl_xor(mx, off, 32));
            float mNew = fmaxf(m_run[r], mx);
            float corr = __expf(m_run[r] - mNew);
            float rs = 0.f;
#pragma unroll
            for (int ni = 0; ni < 4; ++ni) {
                float e = __expf(p[ni][r] - mNew);
                p[ni][r] = e;
                rs += e;
            }
#pragma unroll
            for (int off = 8; off >= 1; off >>= 1)
                rs += __shfl_xor(rs, off, 32);
            l_run[r] = l_run[r] * corr + rs;
            m_run[r] = mNew;
#pragma unroll
            for (int nd = 0; nd < 8; ++nd) o[nd][r] *= corr;
        }

        // P -> LDS (C layout -> row-major bf16)
#pragma unroll
        for (int r = 0; r < 8; ++r) {
            int M = r + 8 * half;
#pragma unroll
            for (int ni = 0; ni < 4; ++ni)
                Ps[wave][M][ni * 16 + l16] = (__bf16)p[ni][r];
        }
        __syncthreads();

        // O += P[16x64] @ V[64x128]
#pragma unroll
        for (int kk2 = 0; kk2 < 2; ++kk2) {
            Frag aP;
#pragma unroll
            for (int v = 0; v < 8; ++v)
                aP.u[v] = *(const unsigned*)&Ps[wave][l16][kk2 * 32 + a_kidx(v, half)];
#pragma unroll
            for (int nd = 0; nd < 8; ++nd) {
                Frag bV;
                int n = nd * 16 + l16;
#pragma unroll
                for (int v = 0; v < 8; ++v)
                    bV.u[v] = *(const unsigned*)&Vts[n][kk2 * 32 + b_kidx(v, half)];
                o[nd] = wmma_bf16(aP.v, bV.v, o[nd]);
            }
        }
    }

#pragma unroll
    for (int nd = 0; nd < 8; ++nd) {
        int col = h * HDIM + nd * 16 + l16;
#pragma unroll
        for (int r = 0; r < 8; ++r) {
            int row = qb * 128 + wave * 16 + r + 8 * half;
            Out[(size_t)row * MDIM + col] = o[nd][r] / l_run[r];
        }
    }
}

extern "C" void kernel_launch(void* const* d_in, const int* in_sizes, int n_in,
                              void* d_out, int out_size, void* d_ws, size_t ws_size,
                              hipStream_t stream) {
    const float* query  = (const float*)d_in[0];
    const float* key_in = (const float*)d_in[1];
    const float* value  = (const float*)d_in[2];
    const float* Wq = (const float*)d_in[3];
    const float* bq = (const float*)d_in[4];
    const float* Wk = (const float*)d_in[5];
    const float* bk = (const float*)d_in[6];
    const float* Wv = (const float*)d_in[7];
    const float* bv = (const float*)d_in[8];
    float* out = (float*)d_out;

    const size_t sm = (size_t)SEQDIM * MDIM;   // 8.39M elems
    const size_t mm = (size_t)MDIM * MDIM;     // 4.19M elems
    __bf16* Xq  = (__bf16*)d_ws;               // converted inputs
    __bf16* Xk  = Xq + sm;
    __bf16* Xv  = Xk + sm;
    __bf16* Wtq = Xv + sm;                     // transposed bf16 weights [N][K]
    __bf16* Wtk = Wtq + mm;
    __bf16* Wtv = Wtk + mm;
    __bf16* Qb  = Wtv + mm;                    // projection outputs
    __bf16* Kb  = Qb + sm;
    __bf16* Vb  = Kb + sm;

    dim3 blk(256);
    // 1) one-time precision/layout conversion
    dim3 gCvt((unsigned)(sm / 1024));
    cvt_bf16<<<gCvt, blk, 0, stream>>>(query,  Xq);
    cvt_bf16<<<gCvt, blk, 0, stream>>>(key_in, Xk);
    cvt_bf16<<<gCvt, blk, 0, stream>>>(value,  Xv);
    dim3 gTr(MDIM / 32, MDIM / 32);
    cvt_transpose_bf16<<<gTr, blk, 0, stream>>>(Wq, Wtq);
    cvt_transpose_bf16<<<gTr, blk, 0, stream>>>(Wk, Wtk);
    cvt_transpose_bf16<<<gTr, blk, 0, stream>>>(Wv, Wtv);

    // 2) projections
    dim3 gGemm(MDIM / 128, SEQDIM / 128);
    gemm_bf16_nt<<<gGemm, blk, 0, stream>>>(Xq, Wtq, bq, Qb);
    gemm_bf16_nt<<<gGemm, blk, 0, stream>>>(Xk, Wtk, bk, Kb);
    gemm_bf16_nt<<<gGemm, blk, 0, stream>>>(Xv, Wtv, bv, Vb);

    // 3) attention
    dim3 gAttn(SEQDIM / 128, NHEAD);
    attn_flash_bf16<<<gAttn, blk, 0, stream>>>(Qb, Kb, Vb, out);
}
